// EViT_71442486001892
// MI455X (gfx1250) — matrix operand order repaired and checked by reading, not compiled, not run on hardware
//
#include <hip/hip_runtime.h>
#include <hip/hip_bf16.h>

// ---------------------------------------------------------------------------
// EViT forward for MI455X (gfx1250, wave32, WMMA).
// GEMMs: C = act(scale * (A_bf16 @ Bt_bf16^T) + bias) via
// v_wmma_f32_16x16x32_bf16. Big GEMMs: 128x128 tile, 8 wmma/K-step/wave.
// Tiles staged global->LDS with GLOBAL_LOAD_ASYNC_TO_LDS_B128 (ASYNCcnt),
// double-buffered, one barrier per K-step, zero VGPR staging.
// ---------------------------------------------------------------------------

typedef __hip_bfloat16 bf16;
typedef __attribute__((ext_vector_type(16))) __bf16 v16bf;
typedef __attribute__((ext_vector_type(8)))  float  v8f;

namespace {
constexpr int kB   = 32;     // batch
constexpr int kD   = 768;    // model dim
constexpr int kH   = 12;     // heads
constexpr int kDh  = 64;     // head dim
constexpr int kL   = 12;     // layers
constexpr int kDFF = 3072;
constexpr int kNC  = 100;
constexpr int kNP  = 196;    // image patches
constexpr int kN1  = 197;    // tokens pre-prune
constexpr int kN1P = 224;    // padded (mult of 32)
constexpr int kN2  = 139;    // tokens post-prune (137 kept + cls + stolen)
constexpr int kN2P = 160;    // padded
constexpr int kSEL = 3;
constexpr int kKeep = 137;   // int(197*0.7)
constexpr int kStolen = 60;  // ranks 136..195 inclusive
}

// Async global->LDS 16B copy (gfx1250): LDS dest offset in a VGPR,
// 64-bit global address in a VGPR pair. Tracked by ASYNCcnt.
__device__ __forceinline__ void async_copy_b128(unsigned ldsOff, const void* gaddr)
{
  asm volatile("global_load_async_to_lds_b128 %0, %1, off"
               :: "v"(ldsOff), "v"((unsigned long long)gaddr)
               : "memory");
}
__device__ __forceinline__ void wait_async0()
{
  asm volatile("s_wait_asynccnt 0x0" ::: "memory");
}

// ----------------------------- GEMM (WMMA) ---------------------------------
// 8 waves as 4(M) x 2(N). Wave tile = (16*AF) x (16*BF).
// Workgroup tile TM x TN = (64*AF) x (32*BF).
template <int AF, int BF>
__global__ __launch_bounds__(256, 1) void gemm_bf16_wmma(
    const bf16* __restrict__ A, const bf16* __restrict__ Bt,
    const float* __restrict__ bias, void* __restrict__ Cout,
    int M, int N, int K,
    long long strA, long long strB, long long strC,
    float scale, int act, int outMode)
{
  constexpr int TM = 64 * AF;
  constexpr int TN = 32 * BF;
  constexpr int AU = TM / 64;       // 16B async copies per thread for A tile
  constexpr int BU = TN / 64;
  __shared__ bf16 shA[2][TM][40];   // 32 K + 8 pad: conflict-free b128 gathers
  __shared__ bf16 shB[2][TN][40];

  const int z = blockIdx.z;
  A  += (long long)z * strA;
  Bt += (long long)z * strB;

  const int tn   = blockIdx.x * TN;
  const int tm   = blockIdx.y * TM;
  const int tid  = threadIdx.x;
  const int lane = tid & 31;
  const int wv   = tid >> 5;        // 0..7
  const int wm   = wv >> 1;         // 0..3 : M strip
  const int wn   = wv & 1;          // 0..1 : N strip
  const int lrow = tid >> 2;        // 0..63 (cooperative tile copies)
  const int lkc  = (tid & 3) << 3;  // 0,8,16,24 bf16 elements

  const int half = lane >> 4;       // 0/1
  const int l15  = lane & 15;
  const int ak   = half << 3;       // A frag: k = (e&7)+16*(e>=8)+8*half
  const int bk   = half << 3;       // B frag: k = (e&7)+16*(e>=8)+8*half

  v8f acc[AF][BF] = {};

  // issue async copies of the k0 K-slab into LDS buffer b (no VGPR staging)
  auto asyncStage = [&](int k0, int b) {
    #pragma unroll
    for (int u = 0; u < AU; ++u) {
      int ra = tm + u * 64 + lrow;
      ra = ra < M ? ra : M - 1;     // clamp: OOB rows masked at store
      async_copy_b128((unsigned)(size_t)&shA[b][u * 64 + lrow][lkc],
                      A + (size_t)ra * K + k0 + lkc);
    }
    #pragma unroll
    for (int u = 0; u < BU; ++u) {
      int rb = tn + u * 64 + lrow;
      rb = rb < N ? rb : N - 1;
      async_copy_b128((unsigned)(size_t)&shB[b][u * 64 + lrow][lkc],
                      Bt + (size_t)rb * K + k0 + lkc);
    }
  };

  // pipeline prologue
  asyncStage(0, 0);
  wait_async0();
  __syncthreads();

  int buf = 0;
  for (int k0 = 0; k0 < K; k0 += 32) {
    const bool more = (k0 + 32) < K;
    if (more) asyncStage(k0 + 32, buf ^ 1);  // DMA overlaps the wmma burst
    if (k0 + 64 < K) {                       // uniform-guard prefetch
      int ra = tm + lrow; ra = ra < M ? ra : M - 1;
      __builtin_prefetch(A + (size_t)ra * K + k0 + 64 + lkc, 0, 1);
    }

    v16bf fa[AF], fb[BF];
    #pragma unroll
    for (int i = 0; i < AF; ++i) {
      uint4* p = (uint4*)&fa[i];
      const int r = wm * (16 * AF) + i * 16 + l15;
      p[0] = *(const uint4*)&shA[buf][r][ak];
      p[1] = *(const uint4*)&shA[buf][r][ak + 16];
    }
    #pragma unroll
    for (int j = 0; j < BF; ++j) {
      uint4* p = (uint4*)&fb[j];
      const int c = wn * (16 * BF) + j * 16 + l15;
      p[0] = *(const uint4*)&shB[buf][c][bk];
      p[1] = *(const uint4*)&shB[buf][c][bk + 16];
    }
    #pragma unroll
    for (int i = 0; i < AF; ++i)
      #pragma unroll
      for (int j = 0; j < BF; ++j)
        acc[i][j] = __builtin_amdgcn_wmma_f32_16x16x32_bf16(
            false, fa[i], false, fb[j], (short)0, acc[i][j], false, false);

    if (more) {
      wait_async0();     // our writes to buf^1 done
      __syncthreads();   // everyone's writes visible / reads of buf done
      buf ^= 1;
    }
  }

  // Epilogue. C/D layout: VGPR v -> row v + 8*half; col = lane&15.
  const int r0 = tm + wm * (16 * AF) + (half << 3);
  const int c0 = tn + wn * (16 * BF) + l15;
  #pragma unroll
  for (int i = 0; i < AF; ++i) {
    #pragma unroll
    for (int j = 0; j < BF; ++j) {
      const int col = c0 + j * 16;
      if (col >= N) continue;
      const float bv = bias ? bias[col] : 0.0f;
      #pragma unroll
      for (int v = 0; v < 8; ++v) {
        const int r = r0 + i * 16 + v;
        if (r >= M) continue;
        float val = acc[i][j][v] * scale + bv;
        if (act == 1) val = 0.5f * val * (1.0f + erff(val * 0.70710678118654752f));
        const long long off = (long long)z * strC + (long long)r * N + col;
        if (outMode == 0)      ((float*)Cout)[off]  = val;
        else if (outMode == 1) ((float*)Cout)[off] += val;  // residual add
        else                   ((bf16*)Cout)[off]  = __float2bfloat16(val);
      }
    }
  }
}

// ------------------------- weight prep kernels -----------------------------
__global__ void convert_f32_bf16(const float* __restrict__ s,
                                 bf16* __restrict__ d, long long n)
{
  for (long long i = (long long)blockIdx.x * blockDim.x + threadIdx.x;
       i < n; i += (long long)gridDim.x * blockDim.x)
    d[i] = __float2bfloat16(s[i]);
}

// src [R][C] f32 -> dst [C][R] bf16 (tiled transpose)
__global__ __launch_bounds__(256) void transpose_f32_bf16(
    const float* __restrict__ src, bf16* __restrict__ dst, int R, int C)
{
  __shared__ float t[32][33];
  const int c0 = blockIdx.x * 32, r0 = blockIdx.y * 32;
  for (int i = threadIdx.y; i < 32; i += 8) {
    const int r = r0 + i, c = c0 + threadIdx.x;
    t[i][threadIdx.x] = (r < R && c < C) ? src[(size_t)r * C + c] : 0.0f;
  }
  __syncthreads();
  for (int i = threadIdx.y; i < 32; i += 8) {
    const int c = c0 + i, r = r0 + threadIdx.x;
    if (c < C && r < R)
      dst[(size_t)c * R + r] = __float2bfloat16(t[threadIdx.x][i]);
  }
}

// ------------------------- patch embed kernels -----------------------------
__global__ void patch_extract(const float* __restrict__ img,
                              bf16* __restrict__ Ap)
{
  const long long total = (long long)kB * kNP * kD;
  for (long long i = (long long)blockIdx.x * blockDim.x + threadIdx.x;
       i < total; i += (long long)gridDim.x * blockDim.x) {
    const int col = (int)(i % kD);
    const long long row = i / kD;
    const int p = (int)(row % kNP);
    const int b = (int)(row / kNP);
    const int c  = col >> 8;
    const int ph = (col >> 4) & 15;
    const int pw = col & 15;
    const int py = p / 14, px = p % 14;
    const float v = img[(((size_t)b * 3 + c) * 224 + py * 16 + ph) * 224 + px * 16 + pw];
    Ap[i] = __float2bfloat16(v);
  }
}

__global__ void assemble_x(const float* __restrict__ Pt,
                           const float* __restrict__ cls,
                           const float* __restrict__ pos,
                           float* __restrict__ x)
{
  const long long total = (long long)kB * kN1 * kD;
  for (long long i = (long long)blockIdx.x * blockDim.x + threadIdx.x;
       i < total; i += (long long)gridDim.x * blockDim.x) {
    const int dd = (int)(i % kD);
    const long long row = i / kD;
    const int n = (int)(row % kN1);
    const int b = (int)(row / kN1);
    const float v = (n == 0) ? cls[dd]
                             : Pt[((size_t)b * kNP + (n - 1)) * kD + dd];
    x[i] = v + pos[(size_t)n * kD + dd];
  }
}

// ------------------------------ layernorm ----------------------------------
__global__ __launch_bounds__(256) void layernorm_kernel(
    const float* __restrict__ x, const float* __restrict__ g,
    const float* __restrict__ b, bf16* __restrict__ out,
    int D, long long rowStride)
{
  const long long row = blockIdx.x;
  const float* xr = x + row * rowStride;
  bf16* orow = out + row * (long long)D;
  float s = 0.f, s2 = 0.f;
  for (int c = threadIdx.x; c < D; c += 256) {
    const float v = xr[c]; s += v; s2 += v * v;
  }
  __shared__ float r1[8], r2[8];
  for (int o = 16; o; o >>= 1) { s += __shfl_xor(s, o, 32); s2 += __shfl_xor(s2, o, 32); }
  const int w = threadIdx.x >> 5;
  if ((threadIdx.x & 31) == 0) { r1[w] = s; r2[w] = s2; }
  __syncthreads();
  float ts = 0.f, ts2 = 0.f;
  for (int i = 0; i < 8; ++i) { ts += r1[i]; ts2 += r2[i]; }
  const float mu  = ts / D;
  const float var = ts2 / D - mu * mu;
  const float inv = rsqrtf(var + 1e-6f);
  for (int c = threadIdx.x; c < D; c += 256)
    orow[c] = __float2bfloat16((xr[c] - mu) * inv * g[c] + b[c]);
}

// ------------------------------ attention ----------------------------------
__global__ void repack_qkv(const bf16* __restrict__ qkv,
                           bf16* __restrict__ q, bf16* __restrict__ k,
                           bf16* __restrict__ vt, int Nt, int Ntp)
{
  const long long total = (long long)kB * kH * Ntp * kDh;
  for (long long i = (long long)blockIdx.x * blockDim.x + threadIdx.x;
       i < total; i += (long long)gridDim.x * blockDim.x) {
    const int d  = (int)(i & 63);
    const int n  = (int)((i >> 6) % Ntp);
    const int bh = (int)(i / ((long long)Ntp * kDh));
    const int h  = bh % kH, b = bh / kH;
    bf16 qv = __float2bfloat16(0.f), kv = qv, vv = qv;
    if (n < Nt) {
      const size_t base = ((size_t)(b * Nt + n)) * (3 * kD) + h * kDh + d;
      qv = qkv[base]; kv = qkv[base + kD]; vv = qkv[base + 2 * kD];
    }
    q[i] = qv; k[i] = kv;
    vt[(size_t)bh * kDh * Ntp + (size_t)d * Ntp + n] = vv;
  }
}

__global__ __launch_bounds__(256) void softmax_kernel(
    const float* __restrict__ S, bf16* __restrict__ P, int Ntp, int Nv)
{
  const long long base = (long long)blockIdx.x * Ntp;
  const int c = threadIdx.x;
  const float v = (c < Nv) ? S[base + c] : -3.0e38f;
  __shared__ float red[8];
  float m = v;
  for (int o = 16; o; o >>= 1) m = fmaxf(m, __shfl_xor(m, o, 32));
  const int w = threadIdx.x >> 5;
  if ((threadIdx.x & 31) == 0) red[w] = m;
  __syncthreads();
  float mx = red[0];
  for (int i = 1; i < 8; ++i) mx = fmaxf(mx, red[i]);
  const float e = (c < Nv) ? __expf(v - mx) : 0.0f;
  float s = e;
  __syncthreads();
  for (int o = 16; o; o >>= 1) s += __shfl_xor(s, o, 32);
  if ((threadIdx.x & 31) == 0) red[w] = s;
  __syncthreads();
  float tot = 0.f;
  for (int i = 0; i < 8; ++i) tot += red[i];
  if (c < Ntp) P[base + c] = __float2bfloat16(e / tot);
}

__global__ void importance_kernel(const bf16* __restrict__ P,
                                  float* __restrict__ imp, int Ntp)
{
  const int b = blockIdx.x;
  const int j = threadIdx.x;
  if (j < kNP) {
    float s = 0.f;
    for (int h = 0; h < kH; ++h)
      s += __bfloat162float(P[((size_t)(b * kH + h) * Ntp + 0) * Ntp + 1 + j]);
    imp[b * kNP + j] = s / (float)kH;
  }
}

__global__ void repack_attnout(const bf16* __restrict__ O,
                               bf16* __restrict__ Oc, int Nt, int Ntp)
{
  const long long total = (long long)kB * Nt * kD;
  for (long long i = (long long)blockIdx.x * blockDim.x + threadIdx.x;
       i < total; i += (long long)gridDim.x * blockDim.x) {
    const int d768 = (int)(i % kD);
    const long long row = i / kD;
    const int n = (int)(row % Nt);
    const int b = (int)(row / Nt);
    const int h = d768 / kDh, d = d768 % kDh;
    Oc[i] = O[((size_t)(b * kH + h) * Ntp + n) * kDh + d];
  }
}

// --------------------------- EViT token select -----------------------------
__global__ __launch_bounds__(256) void select_tokens(
    const float* __restrict__ xold, const float* __restrict__ imp,
    float* __restrict__ xnew)
{
  const int b = blockIdx.x;
  __shared__ int   srcOf[kN2 - 1];            // dest rows 0..137
  __shared__ float sw[64];
  __shared__ int   sidx[64];
  __shared__ int   cnt;
  const float* impb = imp + b * kNP;
  if (threadIdx.x == 0) { cnt = 0; srcOf[0] = 0; }
  __syncthreads();
  const int i = threadIdx.x;
  if (i < kNP) {
    const float vi = impb[i];
    int rank = 0;
    for (int j = 0; j < kNP; ++j) {          // top_k order: ties by lower index
      const float vj = impb[j];
      rank += (vj > vi) || (vj == vi && j < i);
    }
    if (rank < kKeep) srcOf[1 + rank] = 1 + i;
    if (rank >= kKeep - 1) {                 // ranks 136..195 -> stolen set
      const int s = atomicAdd(&cnt, 1);
      sw[s] = vi; sidx[s] = 1 + i;
    }
  }
  __syncthreads();
  const float* xb = xold + (size_t)b * kN1 * kD;
  float* xn = xnew + (size_t)b * kN2 * kD;
  for (int t = threadIdx.x; t < (kN2 - 1) * kD; t += 256) {
    const int r = t / kD, c = t % kD;
    xn[(size_t)r * kD + c] = xb[(size_t)srcOf[r] * kD + c];
  }
  for (int c = threadIdx.x; c < kD; c += 256) {
    float s = 0.f;
    for (int t = 0; t < kStolen; ++t)
      s += sw[t] * xb[(size_t)sidx[t] * kD + c];
    xn[(size_t)(kN2 - 1) * kD + c] = s;
  }
}

// ------------------------------- driver ------------------------------------
static inline size_t alignUp256(size_t v) { return (v + 255) & ~(size_t)255; }

extern "C" void kernel_launch(void* const* d_in, const int* in_sizes, int n_in,
                              void* d_out, int out_size, void* d_ws, size_t ws_size,
                              hipStream_t stream)
{
  (void)in_sizes; (void)n_in; (void)out_size; (void)ws_size;

  const float* in_x      = (const float*)d_in[0];
  const float* patch_w   = (const float*)d_in[1];
  const float* patch_b   = (const float*)d_in[2];
  const float* cls_token = (const float*)d_in[3];
  const float* pos_embed = (const float*)d_in[4];
  const float* ln1_g = (const float*)d_in[5];
  const float* ln1_b = (const float*)d_in[6];
  const float* qkv_w = (const float*)d_in[7];
  const float* qkv_b = (const float*)d_in[8];
  const float* proj_w = (const float*)d_in[9];
  const float* proj_b = (const float*)d_in[10];
  const float* ln2_g = (const float*)d_in[11];
  const float* ln2_b = (const float*)d_in[12];
  const float* fc1_w = (const float*)d_in[13];
  const float* fc1_b = (const float*)d_in[14];
  const float* fc2_w = (const float*)d_in[15];
  const float* fc2_b = (const float*)d_in[16];
  const float* lnf_g = (const float*)d_in[17];
  const float* lnf_b = (const float*)d_in[18];
  const float* head_w = (const float*)d_in[19];
  const float* head_b = (const float*)d_in[20];

  const int M1 = kB * kN1;   // 6304 rows pre-prune
  const int M2 = kB * kN2;   // 4448 rows post-prune
  const int BH = kB * kH;    // 384

  // ---- workspace carve (~266 MB) ----
  char* w = (char*)d_ws;
  auto alloc = [&](size_t bytes) { char* p = w; w += alignUp256(bytes); return p; };
  bf16* wtPatch = (bf16*)alloc((size_t)kD * kD * 2);
  bf16* wtQkv   = (bf16*)alloc((size_t)3 * kD * kD * 2);
  bf16* wtProj  = (bf16*)alloc((size_t)kD * kD * 2);
  bf16* wtFc1   = (bf16*)alloc((size_t)kDFF * kD * 2);
  bf16* wtFc2   = (bf16*)alloc((size_t)kD * kDFF * 2);
  bf16* wtHead  = (bf16*)alloc((size_t)kNC * kD * 2);
  float* x1  = (float*)alloc((size_t)M1 * kD * 4);
  float* x2  = (float*)alloc((size_t)M2 * kD * 4);
  bf16* hbuf = (bf16*)alloc((size_t)M1 * kD * 2);
  bf16* qkvb = (bf16*)alloc((size_t)M1 * 3 * kD * 2);
  bf16* qb   = (bf16*)alloc((size_t)BH * kN1P * kDh * 2);
  bf16* kb   = (bf16*)alloc((size_t)BH * kN1P * kDh * 2);
  bf16* vtb  = (bf16*)alloc((size_t)BH * kN1P * kDh * 2);
  bf16* Ob   = (bf16*)alloc((size_t)BH * kN1P * kDh * 2);
  char* Sreg = alloc((size_t)BH * kN1P * kN1P * 4);  // reused: mid (bf16), Pt (f32)
  bf16* Pbuf = (bf16*)alloc((size_t)BH * kN1P * kN1P * 2);
  bf16* Ocb  = (bf16*)alloc((size_t)M1 * kD * 2);
  bf16* Apat = (bf16*)alloc((size_t)kB * kNP * kD * 2);
  float* impb = (float*)alloc((size_t)kB * kNP * 4);
  bf16* clsb  = (bf16*)alloc((size_t)kB * kD * 2);
  float* Sbuf = (float*)Sreg;
  bf16*  mid  = (bf16*)Sreg;
  float* Pt   = (float*)Sreg;

  const dim3 blk(256);
  // big GEMM: 128x128 tile (8 wmma per K-step per wave)
  auto gemmBig = [&](const bf16* A, const bf16* Bt, const float* bias, void* C,
                     int M, int N, int K, float scale, int act, int mode) {
    dim3 grid((N + 127) / 128, (M + 127) / 128, 1);
    gemm_bf16_wmma<2, 4><<<grid, blk, 0, stream>>>(
        A, Bt, bias, C, M, N, K, 0LL, 0LL, 0LL, scale, act, mode);
  };
  // small / batched GEMM: 64x64 tile
  auto gemmSm = [&](const bf16* A, const bf16* Bt, const float* bias, void* C,
                    int M, int N, int K, long long sA, long long sB, long long sC,
                    float scale, int act, int mode, int Z) {
    dim3 grid((N + 63) / 64, (M + 63) / 64, Z);
    gemm_bf16_wmma<1, 2><<<grid, blk, 0, stream>>>(
        A, Bt, bias, C, M, N, K, sA, sB, sC, scale, act, mode);
  };
  auto transposeW = [&](const float* src, bf16* dst, int R, int C) {
    dim3 grid((C + 31) / 32, (R + 31) / 32, 1);
    transpose_f32_bf16<<<grid, dim3(32, 8), 0, stream>>>(src, dst, R, C);
  };

  // ---- patch embedding ----
  convert_f32_bf16<<<dim3(((size_t)kD * kD + 255) / 256), blk, 0, stream>>>(
      patch_w, wtPatch, (long long)kD * kD);
  patch_extract<<<dim3(((long long)kB * kNP * kD + 255) / 256), blk, 0, stream>>>(
      in_x, Apat);
  gemmBig(Apat, wtPatch, patch_b, Pt, kB * kNP, kD, kD, 1.0f, 0, 0);
  assemble_x<<<dim3(((long long)M1 * kD + 255) / 256), blk, 0, stream>>>(
      Pt, cls_token, pos_embed, x1);
  transposeW(head_w, wtHead, kD, kNC);

  // ---- transformer layers ----
  for (int n = 0; n < kL; ++n) {
    const int Nt1  = (n <= kSEL) ? kN1 : kN2;
    const int Ntp1 = (n <= kSEL) ? kN1P : kN2P;
    const int Ma   = kB * Nt1;
    float* xa = (n <= kSEL) ? x1 : x2;

    transposeW(qkv_w + (size_t)n * kD * 3 * kD, wtQkv, kD, 3 * kD);
    transposeW(proj_w + (size_t)n * kD * kD, wtProj, kD, kD);
    transposeW(fc1_w + (size_t)n * kD * kDFF, wtFc1, kD, kDFF);
    transposeW(fc2_w + (size_t)n * kDFF * kD, wtFc2, kDFF, kD);

    // attention
    layernorm_kernel<<<dim3(Ma), blk, 0, stream>>>(
        xa, ln1_g + n * kD, ln1_b + n * kD, hbuf, kD, (long long)kD);
    gemmBig(hbuf, wtQkv, qkv_b + n * 3 * kD, qkvb, Ma, 3 * kD, kD, 1.0f, 0, 2);
    repack_qkv<<<dim3(((long long)BH * Ntp1 * kDh + 255) / 256), blk, 0, stream>>>(
        qkvb, qb, kb, vtb, Nt1, Ntp1);
    gemmSm(qb, kb, nullptr, Sbuf, Ntp1, Ntp1, kDh,
           (long long)Ntp1 * kDh, (long long)Ntp1 * kDh, (long long)Ntp1 * Ntp1,
           0.125f, 0, 0, BH);
    softmax_kernel<<<dim3(BH * Ntp1), blk, 0, stream>>>(Sbuf, Pbuf, Ntp1, Nt1);
    if (n == kSEL)
      importance_kernel<<<dim3(kB), blk, 0, stream>>>(Pbuf, impb, Ntp1);
    gemmSm(Pbuf, vtb, nullptr, Ob, Ntp1, kDh, Ntp1,
           (long long)Ntp1 * Ntp1, (long long)kDh * Ntp1, (long long)Ntp1 * kDh,
           1.0f, 0, 2, BH);
    repack_attnout<<<dim3(((long long)Ma * kD + 255) / 256), blk, 0, stream>>>(
        Ob, Ocb, Nt1, Ntp1);
    gemmBig(Ocb, wtProj, proj_b + n * kD, xa, Ma, kD, kD, 1.0f, 0, 1);

    // token pruning at SEL (after attn residual, before MLP)
    float* xb = xa;
    int Nt2 = Nt1;
    if (n == kSEL) {
      select_tokens<<<dim3(kB), blk, 0, stream>>>(x1, impb, x2);
      xb = x2; Nt2 = kN2;
    }
    const int Mb = kB * Nt2;

    // MLP
    layernorm_kernel<<<dim3(Mb), blk, 0, stream>>>(
        xb, ln2_g + n * kD, ln2_b + n * kD, hbuf, kD, (long long)kD);
    gemmBig(hbuf, wtFc1, fc1_b + n * kDFF, mid, Mb, kDFF, kD, 1.0f, 1, 2);
    gemmBig(mid, wtFc2, fc2_b + n * kD, xb, Mb, kD, kDFF, 1.0f, 0, 1);
  }

  // ---- final LN on cls rows + head ----
  layernorm_kernel<<<dim3(kB), blk, 0, stream>>>(
      x2, lnf_g, lnf_b, clsb, kD, (long long)kN2 * kD);
  gemmSm(clsb, wtHead, head_b, d_out, kB, kNC, kD, 0, 0, 0, 1.0f, 0, 0, 1);
}
